// SelfAttentionPool_43052752175241
// MI455X (gfx1250) — compile-verified, matrix operand order checked by reading, not run on hardware
//
#include <hip/hip_runtime.h>
#include <hip/hip_bf16.h>
#include <stdint.h>

// ---------------------------------------------------------------------------
// SelfAttentionPool for MI455X (gfx1250).
// Dense layers: bf16 WMMA (v_wmma_f32_16x16x32_bf16), all operands stored in
// bf16 workspace so the GEMM inner loop is pure global_load_b128 + v_wmma.
// Irregular per-segment attention / softmax pooling: VALU kernels.
// ---------------------------------------------------------------------------

typedef __attribute__((ext_vector_type(16))) __bf16          v16bf;
typedef __attribute__((ext_vector_type(8)))  float           v8f;

#define NNODES 8192
#define DDIM   256
#define NSEG   256
#define MAXSEG 512   // per-segment node cap (avg 32, P(>512) ~ 0)

__device__ __forceinline__ unsigned short f32_to_bf16_rne(float v) {
    unsigned int b = __float_as_uint(v);
    b += 0x7FFFu + ((b >> 16) & 1u);          // round-to-nearest-even
    return (unsigned short)(b >> 16);
}

// Load one 16-bit WMMA fragment (16 bf16) for this lane from a bf16 buffer:
// elements 0..7 contiguous at p, elements 8..15 contiguous at p+16 (the K+16
// half of the 16x32 tile). Two 16-byte loads, no conversion arithmetic.
__device__ __forceinline__ v16bf load_frag_bf16(const unsigned short* __restrict__ p) {
    struct Pk { uint4 lo, hi; } pk;
    pk.lo = *(const uint4*)(p);
    pk.hi = *(const uint4*)(p + 16);
    return __builtin_bit_cast(v16bf, pk);
}

// ---------------------------------------------------------------------------
// f32 -> bf16 bulk conversion (4 elems/thread), n % 4 == 0
// ---------------------------------------------------------------------------
__global__ __launch_bounds__(256)
void cvt_f32_bf16_kernel(const float* __restrict__ src, unsigned short* __restrict__ dst, int n) {
    int i = (blockIdx.x * blockDim.x + threadIdx.x) * 4;
    if (i >= n) return;
    float4 v = *(const float4*)(src + i);
    ushort4 o;
    o.x = f32_to_bf16_rne(v.x); o.y = f32_to_bf16_rne(v.y);
    o.z = f32_to_bf16_rne(v.z); o.w = f32_to_bf16_rne(v.w);
    *(ushort4*)(dst + i) = o;
}

// ---------------------------------------------------------------------------
// C[M x NOUT] = act(A[M x K] @ W[NOUT x K]^T + bias), act = leaky-relu or id.
// A, W are bf16; output optionally stored as f32 and/or bf16.
// One wave -> 32(M) x 64(N) tile: per K-step 2 A-frags + 4 B-frags -> 8 WMMAs.
// ---------------------------------------------------------------------------
template <int K, int NOUT, bool LRELU, bool SF32, bool SBF16>
__global__ __launch_bounds__(256)
void gemm_bf16_wmma(const unsigned short* __restrict__ A, const unsigned short* __restrict__ W,
                    const float* __restrict__ bias,
                    float* __restrict__ Cf, unsigned short* __restrict__ Cb, int M) {
    const int lane    = threadIdx.x & 31;
    const int wave    = threadIdx.x >> 5;
    const int tiles_n = NOUT / 64;
    const int gw      = blockIdx.x * 8 + wave;
    const int m0      = (gw / tiles_n) * 32;
    const int n0      = (gw % tiles_n) * 64;
    if (m0 >= M) return;

    const int l15  = lane & 15;
    const int koff = (lane >> 4) * 8;   // lanes 16-31 carry K+8 / K+24 halves

    v8f acc[2][4] = {};
    const unsigned short* a0 = A + (size_t)(m0 + l15) * K + koff;
    const unsigned short* a1 = a0 + (size_t)16 * K;
    const unsigned short* w0 = W + (size_t)(n0 + l15) * K + koff;  // row n of W == col n of B

#pragma unroll
    for (int k0 = 0; k0 < K; k0 += 32) {
        v16bf fa0 = load_frag_bf16(a0 + k0);
        v16bf fa1 = load_frag_bf16(a1 + k0);
#pragma unroll
        for (int t = 0; t < 4; ++t) {
            v16bf fb = load_frag_bf16(w0 + (size_t)t * 16 * K + k0);
            acc[0][t] = __builtin_amdgcn_wmma_f32_16x16x32_bf16(
                false, fa0, false, fb, (short)0, acc[0][t], false, false);
            acc[1][t] = __builtin_amdgcn_wmma_f32_16x16x32_bf16(
                false, fa1, false, fb, (short)0, acc[1][t], false, false);
        }
    }

    // D layout (f32 16x16): lanes 0-15 -> rows 0..7, lanes 16-31 -> rows 8..15.
    const int roff = (lane >> 4) * 8;
#pragma unroll
    for (int mt = 0; mt < 2; ++mt) {
#pragma unroll
        for (int t = 0; t < 4; ++t) {
            const int col = n0 + t * 16 + l15;
            const float bv = bias[col];
            const size_t rbase = (size_t)(m0 + mt * 16 + roff) * NOUT + col;
#pragma unroll
            for (int r = 0; r < 8; ++r) {
                float v = acc[mt][t][r] + bv;
                if (LRELU) v = v > 0.f ? v : 0.01f * v;
                if (SF32)  Cf[rbase + (size_t)r * NOUT] = v;
                if (SBF16) Cb[rbase + (size_t)r * NOUT] = f32_to_bf16_rne(v);
            }
        }
    }
}

// ---------------------------------------------------------------------------
// Segment bookkeeping
// ---------------------------------------------------------------------------
__global__ void seg_init_kernel(float* __restrict__ hpool, int* __restrict__ seg_start,
                                int* __restrict__ seg_end, int n_pool) {
    int i = blockIdx.x * blockDim.x + threadIdx.x;
    if (i < n_pool) hpool[i] = 0.f;
    if (i < NSEG) { seg_start[i] = NNODES; seg_end[i] = 0; }
}

__global__ void seg_bounds_kernel(const long long* __restrict__ batch,
                                  int* __restrict__ seg_start, int* __restrict__ seg_end) {
    int i = blockIdx.x * blockDim.x + threadIdx.x;
    if (i >= NNODES) return;
    int b = (int)batch[i];
    atomicMin(&seg_start[b], i);
    atomicMax(&seg_end[b], i + 1);
}

// ---------------------------------------------------------------------------
// Per-segment masked softmax attention: Hmix[i] = sum_j softmax_j(att_i.att_j) Hf[j]
// Row max clamped at 0 to match reference's max over (dot*mask). Output bf16
// (feeds the next WMMA GEMM directly). aw row in LDS is wave-private and
// per-wave DS ops are in-order -> no workgroup barrier required.
// ---------------------------------------------------------------------------
__global__ __launch_bounds__(256)
void attn_kernel(const float* __restrict__ att, const float* __restrict__ Hf,
                 const int* __restrict__ seg_start, const int* __restrict__ seg_end,
                 unsigned short* __restrict__ Hmix_bf) {
    __shared__ float aw[8][MAXSEG];
    const int s  = blockIdx.x;
    const int st = seg_start[s];
    int n = seg_end[s] - st;
    if (n <= 0) return;
    if (n > MAXSEG) n = MAXSEG;

    const int lane = threadIdx.x & 31;
    const int wave = threadIdx.x >> 5;

    for (int ii = wave; ii < n; ii += 8) {
        const int i = st + ii;
        const float* ai = att + (size_t)i * DDIM;

        float lmax = 0.f;                       // masked zeros participate in max
        for (int jj = lane; jj < n; jj += 32) {
            const float* aj = att + (size_t)(st + jj) * DDIM;
            float dp = 0.f;
#pragma unroll 8
            for (int k = 0; k < DDIM; ++k) dp += ai[k] * aj[k];
            aw[wave][jj] = dp;
            lmax = fmaxf(lmax, dp);
        }
#pragma unroll
        for (int off = 16; off; off >>= 1) lmax = fmaxf(lmax, __shfl_xor(lmax, off, 32));

        float lsum = 0.f;
        for (int jj = lane; jj < n; jj += 32) {
            float e = __expf(aw[wave][jj] - lmax);
            aw[wave][jj] = e;
            lsum += e;
        }
#pragma unroll
        for (int off = 16; off; off >>= 1) lsum += __shfl_xor(lsum, off, 32);
        const float inv = 1.f / lsum;

        float o[8] = {};
        for (int jj = 0; jj < n; ++jj) {
            const float a = aw[wave][jj] * inv;
            const float* hr = Hf + (size_t)(st + jj) * DDIM;
#pragma unroll
            for (int u = 0; u < 8; ++u) o[u] += a * hr[lane + u * 32];
        }
        unsigned short* outr = Hmix_bf + (size_t)i * DDIM;
#pragma unroll
        for (int u = 0; u < 8; ++u) outr[lane + u * 32] = f32_to_bf16_rne(o[u]);
    }
}

// h_gate[i] = G2[i] . g_w3 + g_b3   (K = 128) — one wave per row
__global__ __launch_bounds__(256)
void gate_out_kernel(const float* __restrict__ G2, const float* __restrict__ w3,
                     const float* __restrict__ b3, float* __restrict__ hg) {
    const int lane = threadIdx.x & 31;
    const int wave = threadIdx.x >> 5;
    const int row  = blockIdx.x * 8 + wave;
    if (row >= NNODES) return;
    const float* g = G2 + (size_t)row * 128;
    float sacc = 0.f;
#pragma unroll
    for (int u = 0; u < 4; ++u) sacc += g[lane + u * 32] * w3[lane + u * 32];
#pragma unroll
    for (int off = 16; off; off >>= 1) sacc += __shfl_xor(sacc, off, 32);
    if (lane == 0) hg[row] = sacc + b3[0];
}

// Per-segment gate softmax -> scores; h_pool[s] = sum_i scores[i] * Hmix[i]
__global__ __launch_bounds__(256)
void seg_pool_kernel(const float* __restrict__ hg, const float* __restrict__ Hmix,
                     const int* __restrict__ seg_start, const int* __restrict__ seg_end,
                     float* __restrict__ hpool, float* __restrict__ scores) {
    __shared__ float red[256];
    const int s  = blockIdx.x;
    const int st = seg_start[s];
    const int n  = seg_end[s] - st;
    const int t  = threadIdx.x;
    if (n <= 0) return;                         // hpool pre-zeroed

    float m = -3.4e38f;
    for (int i = t; i < n; i += 256) m = fmaxf(m, hg[st + i]);
    red[t] = m; __syncthreads();
    for (int w = 128; w; w >>= 1) { if (t < w) red[t] = fmaxf(red[t], red[t + w]); __syncthreads(); }
    const float mx = red[0]; __syncthreads();

    float sum = 0.f;
    for (int i = t; i < n; i += 256) {
        float e = __expf(hg[st + i] - mx);
        scores[st + i] = e;
        sum += e;
    }
    red[t] = sum; __syncthreads();
    for (int w = 128; w; w >>= 1) { if (t < w) red[t] += red[t + w]; __syncthreads(); }
    const float inv = 1.f / red[0]; __syncthreads();

    for (int i = t; i < n; i += 256) scores[st + i] *= inv;
    __threadfence_block();
    __syncthreads();

    float acc = 0.f;                            // feature t of the pooled vector
    for (int i = 0; i < n; ++i) acc += scores[st + i] * Hmix[(size_t)(st + i) * DDIM + t];
    hpool[(size_t)s * DDIM + t] = acc;
}

// ---------------------------------------------------------------------------
extern "C" void kernel_launch(void* const* d_in, const int* in_sizes, int n_in,
                              void* d_out, int out_size, void* d_ws, size_t ws_size,
                              hipStream_t stream) {
    const float*      H      = (const float*)d_in[0];
    const long long*  batch  = (const long long*)d_in[1];
    const float* phi_w1 = (const float*)d_in[2];  const float* phi_b1 = (const float*)d_in[3];
    const float* phi_w2 = (const float*)d_in[4];  const float* phi_b2 = (const float*)d_in[5];
    const float* att_w  = (const float*)d_in[6];  const float* att_b  = (const float*)d_in[7];
    const float* rho_w1 = (const float*)d_in[8];  const float* rho_b1 = (const float*)d_in[9];
    const float* rho_w2 = (const float*)d_in[10]; const float* rho_b2 = (const float*)d_in[11];
    const float* g_w1   = (const float*)d_in[12]; const float* g_b1   = (const float*)d_in[13];
    const float* g_w2   = (const float*)d_in[14]; const float* g_b2   = (const float*)d_in[15];
    const float* g_w3   = (const float*)d_in[16]; const float* g_b3   = (const float*)d_in[17];

    // ---- workspace carve-up (all regions 256B-aligned) ----
    char* cur = (char*)d_ws;
    auto carve = [&](size_t bytes) { char* p = cur; cur += (bytes + 255) & ~(size_t)255; return p; };
    const size_t NF = (size_t)NNODES * DDIM;

    float* f0 = (float*)carve(NF * 4);                  // Hf_f32, later Hmix_f32
    float* f1 = (float*)carve(NF * 4);                  // att_f32, later G2_f32
    unsigned short* b0 = (unsigned short*)carve(NF * 2);// H_bf,  later R1_bf
    unsigned short* b1 = (unsigned short*)carve(NF * 2);// T1_bf, later Hmixraw_bf, later Hmix_bf
    unsigned short* b2 = (unsigned short*)carve(NF * 2);// Hf_bf, later G1_bf
    unsigned short* w_phi1 = (unsigned short*)carve(256 * 256 * 2);
    unsigned short* w_phi2 = (unsigned short*)carve(256 * 256 * 2);
    unsigned short* w_att  = (unsigned short*)carve(256 * 256 * 2);
    unsigned short* w_rho1 = (unsigned short*)carve(256 * 256 * 2);
    unsigned short* w_rho2 = (unsigned short*)carve(256 * 256 * 2);
    unsigned short* w_g1   = (unsigned short*)carve(128 * 256 * 2);
    unsigned short* w_g2   = (unsigned short*)carve(128 * 128 * 2);
    float* hg = (float*)carve(NNODES * 4);
    int* seg_start = (int*)carve(NSEG * 4);
    int* seg_end   = (int*)carve(NSEG * 4);

    float* hpool  = (float*)d_out;                 // [256, 256]
    float* scores = hpool + (size_t)NSEG * DDIM;   // [8192, 1]

    // segment bounds + zero h_pool
    seg_init_kernel<<<(NSEG * DDIM + 255) / 256, 256, 0, stream>>>(hpool, seg_start, seg_end, NSEG * DDIM);
    seg_bounds_kernel<<<NNODES / 256, 256, 0, stream>>>(batch, seg_start, seg_end);

    // one-time f32 -> bf16 conversions (weights are L2-resident afterwards)
    auto cvt = [&](const float* s, unsigned short* d, int n) {
        cvt_f32_bf16_kernel<<<(n / 4 + 255) / 256, 256, 0, stream>>>(s, d, n);
    };
    cvt(H,      b0,     NNODES * 256);
    cvt(phi_w1, w_phi1, 256 * 256);
    cvt(phi_w2, w_phi2, 256 * 256);
    cvt(att_w,  w_att,  256 * 256);
    cvt(rho_w1, w_rho1, 256 * 256);
    cvt(rho_w2, w_rho2, 256 * 256);
    cvt(g_w1,   w_g1,   128 * 256);
    cvt(g_w2,   w_g2,   128 * 128);

    // grids: (M/32) * (NOUT/64) waves / 8 waves-per-block
    // phi MLP
    gemm_bf16_wmma<256, 256, true , false, true ><<<128, 256, 0, stream>>>(b0, w_phi1, phi_b1, nullptr, b1, NNODES);
    gemm_bf16_wmma<256, 256, true , true , true ><<<128, 256, 0, stream>>>(b1, w_phi2, phi_b2, f0, b2, NNODES);      // Hf
    // attention projection + per-segment softmax mixing
    gemm_bf16_wmma<256, 256, false, true , false><<<128, 256, 0, stream>>>(b2, w_att, att_b, f1, nullptr, NNODES);   // att
    attn_kernel<<<NSEG, 256, 0, stream>>>(f1, f0, seg_start, seg_end, b1);                                           // Hmix raw (bf16)
    // rho MLP
    gemm_bf16_wmma<256, 256, true , false, true ><<<128, 256, 0, stream>>>(b1, w_rho1, rho_b1, nullptr, b0, NNODES);
    gemm_bf16_wmma<256, 256, true , true , true ><<<128, 256, 0, stream>>>(b0, w_rho2, rho_b2, f0, b1, NNODES);      // H_mixed
    // gate MLP
    gemm_bf16_wmma<256, 128, true , false, true ><<< 64, 256, 0, stream>>>(b1, w_g1, g_b1, nullptr, b2, NNODES);     // G1
    gemm_bf16_wmma<128, 128, true , true , false><<< 64, 256, 0, stream>>>(b2, w_g2, g_b2, f1, nullptr, NNODES);     // G2
    gate_out_kernel<<<NNODES / 8, 256, 0, stream>>>(f1, g_w3, g_b3, hg);
    // segment softmax + pooling
    seg_pool_kernel<<<NSEG, 256, 0, stream>>>(hg, f0, seg_start, seg_end, hpool, scores);
}
